// CrossAttention3D_50019189129546
// MI455X (gfx1250) — matrix operand order repaired and checked by reading, not compiled, not run on hardware
//
#include <hip/hip_runtime.h>
#include <hip/hip_bf16.h>
#include <math.h>

#define DIMC   512
#define NHEADS 8
#define DH     64
#define NTQ    8192
#define NTK    2048
#define EPSV   1e-5f
#define SCALE  0.125f   // 64^-0.5

typedef __bf16 bf16_t;
typedef __bf16 v16bf __attribute__((ext_vector_type(16)));
typedef __bf16 v8bf  __attribute__((ext_vector_type(8)));
typedef float  v8f   __attribute__((ext_vector_type(8)));
typedef unsigned int u32x4 __attribute__((ext_vector_type(4)));
typedef int          i32x4 __attribute__((ext_vector_type(4)));
typedef int          i32x8 __attribute__((ext_vector_type(8)));

#if defined(__has_builtin)
#if __has_builtin(__builtin_amdgcn_tensor_load_to_lds)
#define HAVE_TDM 1
#endif
#endif
#ifndef HAVE_TDM
#define HAVE_TDM 0
#endif

union FragU { v16bf v; v8bf h[2]; };

// ---------------------------------------------------------------------------
// WMMA fragment helpers (layouts per CDNA5 ISA 7.12.2, wave32)
// ---------------------------------------------------------------------------

// A operand, 16(M) x 32(K), row-major source.
// lanes 0-15: row=lane, K = {0..7, 16..23}; lanes 16-31: row=lane-16, K = {8..15, 24..31}
__device__ __forceinline__ v16bf frag_a_rowmajor(const bf16_t* base, int ld, int lane) {
  const int r  = lane & 15;
  const int kh = (lane >> 4) << 3;          // 0 or 8
  const bf16_t* p = base + r * ld + kh;
  FragU u;
  u.h[0] = *(const v8bf*)(p);               // K = kh .. kh+7
  u.h[1] = *(const v8bf*)(p + 16);          // K = kh+16 .. kh+23
  return u.v;
}

// B operand, 32(K) x 16(N). Source is B^T row-major (row n holds K contiguous).
// lanes 0-15: col=lane, K = {0..7, 16..23}; lanes 16-31: col=lane-16, K = {8..15, 24..31}
__device__ __forceinline__ v16bf frag_b_rowmajor(const bf16_t* base, int ld, int lane) {
  const int n  = lane & 15;
  const int kh = (lane >> 4) << 3;          // 0 or 8
  const bf16_t* p = base + n * ld + kh;
  FragU u;
  u.h[0] = *(const v8bf*)(p);
  u.h[1] = *(const v8bf*)(p + 16);
  return u.v;
}

__device__ __forceinline__ v8f wmma_bf16(v16bf a, v16bf b, v8f c) {
  return __builtin_amdgcn_wmma_f32_16x16x32_bf16(false, a, false, b, (short)0, c,
                                                 false, false);
}

__device__ __forceinline__ float gelu_exact(float x) {
  return 0.5f * x * (1.0f + erff(x * 0.70710678118654752f));
}

// ---------------------------------------------------------------------------
// Tensor Data Mover: 2D tile (bf16) global -> LDS with row padding.
// D# packing per CDNA5 ISA 8.3/8.4. All inputs are wave-uniform -> SGPRs.
//   pad_interval_code: rows of 2^(code+1) DWORDs before each pad insertion
//   pad_amount_code:   (code+1) DWORDs of LDS padding
// ---------------------------------------------------------------------------
#if HAVE_TDM
__device__ __forceinline__ void tdm_load_tile_bf16(
    unsigned lds_byte_addr, const void* gtile,
    unsigned tile_w, unsigned tile_h,
    unsigned long long row_stride_elems, unsigned tensor_h,
    unsigned pad_interval_code, unsigned pad_amount_code) {
  unsigned long long ga = (unsigned long long)gtile;
  u32x4 g0;
  g0[0] = 1u;                                        // count=1, user descriptor
  g0[1] = lds_byte_addr;                             // lds_addr
  g0[2] = (unsigned)(ga & 0xffffffffu);              // global_addr[31:0]
  g0[3] = (unsigned)((ga >> 32) & 0x01ffffffu) | (2u << 30);  // addr[56:32] | type=2

  unsigned long long td0 = row_stride_elems;         // tensor_dim0 (elements)
  unsigned long long s0  = row_stride_elems;         // tensor_dim0_stride
  unsigned w0 = (1u << 16)                           // data_size = 1 -> 2 bytes
              | (1u << 20)                           // pad_enable
              | (pad_interval_code << 22)
              | (pad_amount_code << 25);
  unsigned w1 = (unsigned)(td0 & 0xffffu) << 16;     // [15:0] atomic addr=0, [31:16] td0 lo
  unsigned w2 = (unsigned)((td0 >> 16) & 0xffffu) | ((tensor_h & 0xffffu) << 16);
  unsigned w3 = ((tensor_h >> 16) & 0xffffu) | ((tile_w & 0xffffu) << 16);
  unsigned w4 = (tile_h & 0xffffu);                  // tile_dim1; tile_dim2 = 0
  unsigned w5 = (unsigned)(s0 & 0xffffffffu);
  unsigned w6 = (unsigned)((s0 >> 32) & 0xffffu);    // dim1_stride = 0 (2D)
  unsigned w7 = 0u;
  i32x8 g1 = { (int)w0, (int)w1, (int)w2, (int)w3,
               (int)w4, (int)w5, (int)w6, (int)w7 };
  i32x4 z4 = { 0, 0, 0, 0 };
#if __clang_major__ >= 23
  i32x8 z8 = { 0, 0, 0, 0, 0, 0, 0, 0 };
  __builtin_amdgcn_tensor_load_to_lds(g0, g1, z4, z4, z8, 0);
#else
  __builtin_amdgcn_tensor_load_to_lds(g0, g1, z4, z4, 0);
#endif
}
#endif

__device__ __forceinline__ unsigned lds_offset_of(const void* p) {
  // LDS aperture maps flat addr[31:0] directly to the LDS byte address.
  return (unsigned)(unsigned long long)p;
}

// ---------------------------------------------------------------------------
// Weight convert: W (In x Out, f32) -> W^T (Out x In, bf16)
// ---------------------------------------------------------------------------
__global__ __launch_bounds__(256) void convert_wT_kernel(const float* __restrict__ W,
                                                         bf16_t* __restrict__ WT,
                                                         int In, int Out) {
  int idx = blockIdx.x * blockDim.x + threadIdx.x;
  if (idx >= In * Out) return;
  int r = idx / Out;
  int c = idx - r * Out;
  WT[(size_t)c * In + r] = (bf16_t)W[idx];
}

// ---------------------------------------------------------------------------
// LayerNorm over channels for channel-major input (c x Ntok) -> token-major bf16
// ---------------------------------------------------------------------------
__global__ __launch_bounds__(256) void ln_cmajor_kernel(const float* __restrict__ X,
                                                        const float* __restrict__ g,
                                                        const float* __restrict__ b,
                                                        bf16_t* __restrict__ Y,
                                                        int Ntok) {
  int t = blockIdx.x * blockDim.x + threadIdx.x;
  if (t >= Ntok) return;
  float s = 0.f, ss = 0.f;
  for (int c = 0; c < DIMC; ++c) {
    float v = X[(size_t)c * Ntok + t];
    s += v; ss += v * v;
  }
  float mu  = s * (1.0f / DIMC);
  float var = ss * (1.0f / DIMC) - mu * mu;
  float rs  = rsqrtf(var + EPSV);
  for (int c = 0; c < DIMC; ++c) {
    float v = (X[(size_t)c * Ntok + t] - mu) * rs * g[c] + b[c];
    Y[(size_t)t * DIMC + c] = (bf16_t)v;
  }
}

// ---------------------------------------------------------------------------
// Row LayerNorm (token-major f32 input), wave32-per-row.
// ---------------------------------------------------------------------------
__global__ __launch_bounds__(256) void row_ln_kernel(const float* __restrict__ X,
                                                     const float* __restrict__ g,
                                                     const float* __restrict__ bta,
                                                     float* __restrict__ Yf,
                                                     bf16_t* __restrict__ Yb,
                                                     int Ntok) {
  int wid = threadIdx.x >> 5, lane = threadIdx.x & 31;
  int row = blockIdx.x * (blockDim.x >> 5) + wid;
  if (row >= Ntok) return;
  const float* x = X + (size_t)row * DIMC;
  float v[16];
  float s = 0.f, ss = 0.f;
#pragma unroll
  for (int i = 0; i < 16; ++i) {
    float t = x[lane + i * 32];
    v[i] = t; s += t; ss += t * t;
  }
#pragma unroll
  for (int m = 1; m < 32; m <<= 1) { s += __shfl_xor(s, m, 32); ss += __shfl_xor(ss, m, 32); }
  float mu  = s * (1.0f / DIMC);
  float var = ss * (1.0f / DIMC) - mu * mu;
  float rs  = rsqrtf(var + EPSV);
#pragma unroll
  for (int i = 0; i < 16; ++i) {
    int c = lane + i * 32;
    float y = (v[i] - mu) * rs * g[c] + bta[c];
    if (Yf) Yf[(size_t)row * DIMC + c] = y;
    if (Yb) Yb[(size_t)row * DIMC + c] = (bf16_t)y;
  }
}

// Final: out[c * Ntok + row] = LN(Z[row] + H[row]) (transposed, channel-major f32)
__global__ __launch_bounds__(256) void final_ln_transpose_kernel(const float* __restrict__ Z,
                                                                 const float* __restrict__ H,
                                                                 const float* __restrict__ g,
                                                                 const float* __restrict__ bta,
                                                                 float* __restrict__ out,
                                                                 int Ntok) {
  int wid = threadIdx.x >> 5, lane = threadIdx.x & 31;
  int row = blockIdx.x * (blockDim.x >> 5) + wid;
  if (row >= Ntok) return;
  const float* z = Z + (size_t)row * DIMC;
  const float* h = H + (size_t)row * DIMC;
  float v[16];
  float s = 0.f, ss = 0.f;
#pragma unroll
  for (int i = 0; i < 16; ++i) {
    float t = z[lane + i * 32] + h[lane + i * 32];
    v[i] = t; s += t; ss += t * t;
  }
#pragma unroll
  for (int m = 1; m < 32; m <<= 1) { s += __shfl_xor(s, m, 32); ss += __shfl_xor(ss, m, 32); }
  float mu  = s * (1.0f / DIMC);
  float var = ss * (1.0f / DIMC) - mu * mu;
  float rs  = rsqrtf(var + EPSV);
#pragma unroll
  for (int i = 0; i < 16; ++i) {
    int c = lane + i * 32;
    out[(size_t)c * Ntok + row] = (v[i] - mu) * rs * g[c] + bta[c];
  }
}

// ---------------------------------------------------------------------------
// Tiled WMMA GEMM: C[M x N] = A[M x K] * B[K x N] + bias, B given as B^T (N x K).
// Block tile 64 x 128, 256 threads = 8 waves (2 x 4), each wave 32x32 (4 frags).
// A/B tiles staged by the Tensor Data Mover (wave 0 issues, TENSORcnt + barrier
// publishes); rows padded 64B+16B -> LDS stride 40 elements.
// ---------------------------------------------------------------------------
#define GLDA 40
#define GLDB 40

__global__ __launch_bounds__(256) void gemm_bf16_wmma_kernel(
    const bf16_t* __restrict__ A, const bf16_t* __restrict__ BT,
    const float* __restrict__ bias, float* __restrict__ Cf, bf16_t* __restrict__ Cb,
    int M, int N, int K, int do_gelu) {
  __shared__ bf16_t lds_a[64 * GLDA];
  __shared__ bf16_t lds_b[128 * GLDB];

  const int tid  = threadIdx.x;
  const int lane = tid & 31;
  const int wid  = tid >> 5;
  const int wm   = (wid & 1) * 32;
  const int wn   = (wid >> 1) * 32;
  const int m0   = blockIdx.y * 64;
  const int n0   = blockIdx.x * 128;

  v8f c00 = {}, c01 = {}, c10 = {}, c11 = {};

  const int arow   = tid >> 2;          // 0..63
  const int achunk = (tid & 3) * 8;     // 0,8,16,24

  for (int k0 = 0; k0 < K; k0 += 32) {
    if (k0 + 32 < K) {
      __builtin_prefetch(A  + (size_t)(m0 + arow) * K + k0 + 32 + achunk, 0, 1);
      __builtin_prefetch(BT + (size_t)(n0 + arow) * K + k0 + 32 + achunk, 0, 1);
      __builtin_prefetch(BT + (size_t)(n0 + arow + 64) * K + k0 + 32 + achunk, 0, 1);
    }
#if HAVE_TDM
    if (wid == 0) {
      // 32-elem rows (64B) + 16B pad per row => interval code 3, amount code 3
      tdm_load_tile_bf16(lds_offset_of(&lds_a[0]), A + (size_t)m0 * K + k0,
                         32, 64, (unsigned long long)K, 64u, 3u, 3u);
      tdm_load_tile_bf16(lds_offset_of(&lds_b[0]), BT + (size_t)n0 * K + k0,
                         32, 128, (unsigned long long)K, 128u, 3u, 3u);
      __builtin_amdgcn_s_wait_tensorcnt(0);
    }
    __syncthreads();
#else
    *(v8bf*)&lds_a[arow * GLDA + achunk] =
        *(const v8bf*)&A[(size_t)(m0 + arow) * K + k0 + achunk];
    *(v8bf*)&lds_b[arow * GLDB + achunk] =
        *(const v8bf*)&BT[(size_t)(n0 + arow) * K + k0 + achunk];
    *(v8bf*)&lds_b[(arow + 64) * GLDB + achunk] =
        *(const v8bf*)&BT[(size_t)(n0 + arow + 64) * K + k0 + achunk];
    __syncthreads();
#endif

    v16bf a0 = frag_a_rowmajor(&lds_a[(wm)      * GLDA], GLDA, lane);
    v16bf a1 = frag_a_rowmajor(&lds_a[(wm + 16) * GLDA], GLDA, lane);
    v16bf b0 = frag_b_rowmajor(&lds_b[(wn)      * GLDB], GLDB, lane);
    v16bf b1 = frag_b_rowmajor(&lds_b[(wn + 16) * GLDB], GLDB, lane);

    c00 = wmma_bf16(a0, b0, c00);
    c01 = wmma_bf16(a0, b1, c01);
    c10 = wmma_bf16(a1, b0, c10);
    c11 = wmma_bf16(a1, b1, c11);
    __syncthreads();
  }

  const int nn = lane & 15;
  const int mb = (lane >> 4) * 8;
  const int gc0 = n0 + wn + nn;
  const int gc1 = gc0 + 16;
  const float bb0 = bias ? bias[gc0] : 0.f;
  const float bb1 = bias ? bias[gc1] : 0.f;
#pragma unroll
  for (int r = 0; r < 8; ++r) {
    int gr0 = m0 + wm + mb + r;
    int gr1 = gr0 + 16;
    float v00 = c00[r] + bb0, v01 = c01[r] + bb1;
    float v10 = c10[r] + bb0, v11 = c11[r] + bb1;
    if (do_gelu) {
      v00 = gelu_exact(v00); v01 = gelu_exact(v01);
      v10 = gelu_exact(v10); v11 = gelu_exact(v11);
    }
    if (Cf) {
      Cf[(size_t)gr0 * N + gc0] = v00; Cf[(size_t)gr0 * N + gc1] = v01;
      Cf[(size_t)gr1 * N + gc0] = v10; Cf[(size_t)gr1 * N + gc1] = v11;
    }
    if (Cb) {
      Cb[(size_t)gr0 * N + gc0] = (bf16_t)v00; Cb[(size_t)gr0 * N + gc1] = (bf16_t)v01;
      Cb[(size_t)gr1 * N + gc0] = (bf16_t)v10; Cb[(size_t)gr1 * N + gc1] = (bf16_t)v11;
    }
  }
}

// ---------------------------------------------------------------------------
// Flash attention. Block = 128 threads = 4 waves; each wave owns 16 Q rows.
// K tile DMA'd by the TDM (wave 0) while all waves stage V transposed in LDS;
// online softmax in registers; P re-laid out through per-wave LDS scratch.
// ---------------------------------------------------------------------------
#define LDK  72   // 64 d + 8 pad (144B rows, 16B aligned)
#define LDVT 40   // 32 k + 8 pad
#define LDP  40   // 32 n + 8 pad

__global__ __launch_bounds__(128) void attention_kernel(
    const bf16_t* __restrict__ qh, const bf16_t* __restrict__ kh,
    const bf16_t* __restrict__ vh, bf16_t* __restrict__ aout) {
  __shared__ bf16_t lds_k[32 * LDK];
  __shared__ bf16_t lds_vt[64 * LDVT];
  __shared__ bf16_t lds_p[4 * 16 * LDP];

  const int tid  = threadIdx.x;
  const int lane = tid & 31;
  const int wid  = tid >> 5;
  const int head = blockIdx.y;
  const int q0row = blockIdx.x * 64 + wid * 16;

  const bf16_t* qbase = qh + (size_t)q0row * DIMC + head * DH;
  v16bf qf0 = frag_a_rowmajor(qbase, DIMC, lane);       // d = 0..31
  v16bf qf1 = frag_a_rowmajor(qbase + 32, DIMC, lane);  // d = 32..63

  v8f o0 = {}, o1 = {}, o2 = {}, o3 = {};
  float mrow[8], lrow[8];
#pragma unroll
  for (int r = 0; r < 8; ++r) { mrow[r] = -1e30f; lrow[r] = 0.f; }

  bf16_t* pw = &lds_p[wid * 16 * LDP];
  const int mb = (lane >> 4) * 8;
  const int nn = lane & 15;

  for (int kb = 0; kb < NTK; kb += 32) {
#if HAVE_TDM
    if (wid == 0) {
      // 64-elem rows (128B) + 16B pad => interval code 4, amount code 3
      tdm_load_tile_bf16(lds_offset_of(&lds_k[0]),
                         kh + (size_t)kb * DIMC + head * DH,
                         64, 32, (unsigned long long)DIMC, 32u, 4u, 3u);
    }
#endif
    // All waves stage V transposed (d-major) while the K DMA is in flight.
#pragma unroll
    for (int it = 0; it < 2; ++it) {
      int cid = tid + it * 128;       // 0..255
      int kr  = cid >> 3;             // 0..31
      int dc  = (cid & 7) * 8;        // 0..56
#if !HAVE_TDM
      *(v8bf*)&lds_k[kr * LDK + dc] =
          *(const v8bf*)(kh + (size_t)(kb + kr) * DIMC + head * DH + dc);
#endif
      v8bf vv = *(const v8bf*)(vh + (size_t)(kb + kr) * DIMC + head * DH + dc);
#pragma unroll
      for (int j = 0; j < 8; ++j) lds_vt[(dc + j) * LDVT + kr] = vv[j];
    }
#if HAVE_TDM
    if (wid == 0) __builtin_amdgcn_s_wait_tensorcnt(0);
#endif
    __syncthreads();

    // S = scale * Q Kt : two 16-col subtiles, contraction d=64 (two k=32 steps)
    v8f s0 = {}, s1 = {};
    s0 = wmma_bf16(qf0, frag_b_rowmajor(&lds_k[0], LDK, lane), s0);
    s0 = wmma_bf16(qf1, frag_b_rowmajor(&lds_k[32], LDK, lane), s0);
    s1 = wmma_bf16(qf0, frag_b_rowmajor(&lds_k[16 * LDK], LDK, lane), s1);
    s1 = wmma_bf16(qf1, frag_b_rowmajor(&lds_k[16 * LDK + 32], LDK, lane), s1);

    // Online softmax (row r lives in one 16-lane half; reduce with xor 1,2,4,8)
#pragma unroll
    for (int r = 0; r < 8; ++r) {
      float a0 = s0[r] * SCALE;
      float a1 = s1[r] * SCALE;
      float mx = fmaxf(a0, a1);
#pragma unroll
      for (int m = 1; m < 16; m <<= 1) mx = fmaxf(mx, __shfl_xor(mx, m, 32));
      float mn = fmaxf(mrow[r], mx);
      float alpha = __expf(mrow[r] - mn);
      float p0 = __expf(a0 - mn);
      float p1 = __expf(a1 - mn);
      float rs = p0 + p1;
#pragma unroll
      for (int m = 1; m < 16; m <<= 1) rs += __shfl_xor(rs, m, 32);
      lrow[r] = lrow[r] * alpha + rs;
      mrow[r] = mn;
      o0[r] *= alpha; o1[r] *= alpha; o2[r] *= alpha; o3[r] *= alpha;
      pw[(mb + r) * LDP + nn]      = (bf16_t)p0;
      pw[(mb + r) * LDP + 16 + nn] = (bf16_t)p1;
    }

    // O += P (16x32) * V (32x64)
    v16bf pf = frag_a_rowmajor(pw, LDP, lane);
    o0 = wmma_bf16(pf, frag_b_rowmajor(&lds_vt[0 * 16 * LDVT], LDVT, lane), o0);
    o1 = wmma_bf16(pf, frag_b_rowmajor(&lds_vt[1 * 16 * LDVT], LDVT, lane), o1);
    o2 = wmma_bf16(pf, frag_b_rowmajor(&lds_vt[2 * 16 * LDVT], LDVT, lane), o2);
    o3 = wmma_bf16(pf, frag_b_rowmajor(&lds_vt[3 * 16 * LDVT], LDVT, lane), o3);
    __syncthreads();
  }

  bf16_t* ob = aout + (size_t)q0row * DIMC + head * DH;
#pragma unroll
  for (int r = 0; r < 8; ++r) {
    float inv = 1.0f / lrow[r];
    size_t ro = (size_t)(mb + r) * DIMC;
    ob[ro + nn]      = (bf16_t)(o0[r] * inv);
    ob[ro + 16 + nn] = (bf16_t)(o1[r] * inv);
    ob[ro + 32 + nn] = (bf16_t)(o2[r] * inv);
    ob[ro + 48 + nn] = (bf16_t)(o3[r] * inv);
  }
}

// ---------------------------------------------------------------------------
// Host side
// ---------------------------------------------------------------------------
extern "C" void kernel_launch(void* const* d_in, const int* in_sizes, int n_in,
                              void* d_out, int out_size, void* d_ws, size_t ws_size,
                              hipStream_t stream) {
  const float* q       = (const float*)d_in[0];
  const float* k       = (const float*)d_in[1];
  const float* v       = (const float*)d_in[2];
  const float* ln_q_g  = (const float*)d_in[3];
  const float* ln_q_b  = (const float*)d_in[4];
  const float* ln_k_g  = (const float*)d_in[5];
  const float* ln_k_b  = (const float*)d_in[6];
  const float* ln_v_g  = (const float*)d_in[7];
  const float* ln_v_b  = (const float*)d_in[8];
  const float* Wq      = (const float*)d_in[9];
  const float* bq      = (const float*)d_in[10];
  const float* Wk      = (const float*)d_in[11];
  const float* bk      = (const float*)d_in[12];
  const float* Wv      = (const float*)d_in[13];
  const float* bv      = (const float*)d_in[14];
  const float* Wp      = (const float*)d_in[15];
  const float* bp      = (const float*)d_in[16];
  const float* ln_pre_g  = (const float*)d_in[17];
  const float* ln_pre_b  = (const float*)d_in[18];
  const float* W1      = (const float*)d_in[19];
  const float* b1      = (const float*)d_in[20];
  const float* W2      = (const float*)d_in[21];
  const float* b2      = (const float*)d_in[22];
  const float* ln_post_g = (const float*)d_in[23];
  const float* ln_post_b = (const float*)d_in[24];
  float* out = (float*)d_out;

  char* ws = (char*)d_ws;
  size_t off = 0;
  auto take = [&](size_t bytes) -> char* {
    char* p = ws + off;
    off = (off + bytes + 255) & ~(size_t)255;
    return p;
  };

  bf16_t* Xq  = (bf16_t*)take((size_t)NTQ * DIMC * 2);
  bf16_t* Xk  = (bf16_t*)take((size_t)NTK * DIMC * 2);
  bf16_t* Xv  = (bf16_t*)take((size_t)NTK * DIMC * 2);
  bf16_t* WqT = (bf16_t*)take((size_t)DIMC * DIMC * 2);
  bf16_t* WkT = (bf16_t*)take((size_t)DIMC * DIMC * 2);
  bf16_t* WvT = (bf16_t*)take((size_t)DIMC * DIMC * 2);
  bf16_t* WpT = (bf16_t*)take((size_t)DIMC * DIMC * 2);
  bf16_t* W1T = (bf16_t*)take((size_t)DIMC * 2 * DIMC * 2);
  bf16_t* W2T = (bf16_t*)take((size_t)DIMC * 2 * DIMC * 2);
  bf16_t* qhb = (bf16_t*)take((size_t)NTQ * DIMC * 2);
  bf16_t* khb = (bf16_t*)take((size_t)NTK * DIMC * 2);
  bf16_t* vhb = (bf16_t*)take((size_t)NTK * DIMC * 2);
  bf16_t* ab  = (bf16_t*)take((size_t)NTQ * DIMC * 2);
  float*  z0  = (float*)take((size_t)NTQ * DIMC * 4);
  float*  znf = (float*)take((size_t)NTQ * DIMC * 4);
  bf16_t* znb = (bf16_t*)take((size_t)NTQ * DIMC * 2);
  bf16_t* h1  = (bf16_t*)take((size_t)NTQ * 2 * DIMC * 2);
  float*  h2  = z0;  // z0 dead after pre-LN; reuse for MLP output

  // 1) convert weights -> bf16 transposed
  {
    int n = DIMC * DIMC;
    convert_wT_kernel<<<(n + 255) / 256, 256, 0, stream>>>(Wq, WqT, DIMC, DIMC);
    convert_wT_kernel<<<(n + 255) / 256, 256, 0, stream>>>(Wk, WkT, DIMC, DIMC);
    convert_wT_kernel<<<(n + 255) / 256, 256, 0, stream>>>(Wv, WvT, DIMC, DIMC);
    convert_wT_kernel<<<(n + 255) / 256, 256, 0, stream>>>(Wp, WpT, DIMC, DIMC);
    int n2 = DIMC * 2 * DIMC;
    convert_wT_kernel<<<(n2 + 255) / 256, 256, 0, stream>>>(W1, W1T, DIMC, 2 * DIMC);
    convert_wT_kernel<<<(n2 + 255) / 256, 256, 0, stream>>>(W2, W2T, 2 * DIMC, DIMC);
  }

  // 2) LayerNorms on channel-major inputs -> token-major bf16
  ln_cmajor_kernel<<<NTQ / 256, 256, 0, stream>>>(q, ln_q_g, ln_q_b, Xq, NTQ);
  ln_cmajor_kernel<<<NTK / 256, 256, 0, stream>>>(k, ln_k_g, ln_k_b, Xk, NTK);
  ln_cmajor_kernel<<<NTK / 256, 256, 0, stream>>>(v, ln_v_g, ln_v_b, Xv, NTK);

  // 3) Q/K/V projections (WMMA GEMM)
  gemm_bf16_wmma_kernel<<<dim3(DIMC / 128, NTQ / 64), 256, 0, stream>>>(
      Xq, WqT, bq, nullptr, qhb, NTQ, DIMC, DIMC, 0);
  gemm_bf16_wmma_kernel<<<dim3(DIMC / 128, NTK / 64), 256, 0, stream>>>(
      Xk, WkT, bk, nullptr, khb, NTK, DIMC, DIMC, 0);
  gemm_bf16_wmma_kernel<<<dim3(DIMC / 128, NTK / 64), 256, 0, stream>>>(
      Xv, WvT, bv, nullptr, vhb, NTK, DIMC, DIMC, 0);

  // 4) Flash attention (WMMA + TDM)
  attention_kernel<<<dim3(NTQ / 64, NHEADS), 128, 0, stream>>>(qhb, khb, vhb, ab);

  // 5) Output projection -> z0 (f32)
  gemm_bf16_wmma_kernel<<<dim3(DIMC / 128, NTQ / 64), 256, 0, stream>>>(
      ab, WpT, bp, z0, nullptr, NTQ, DIMC, DIMC, 0);

  // 6) pre-LN -> znf (f32 residual) + znb (bf16 GEMM input)
  row_ln_kernel<<<NTQ / 8, 256, 0, stream>>>(z0, ln_pre_g, ln_pre_b, znf, znb, NTQ);

  // 7) MLP: h1 = gelu(zn @ W1 + b1) ; h2 = h1 @ W2 + b2
  gemm_bf16_wmma_kernel<<<dim3(2 * DIMC / 128, NTQ / 64), 256, 0, stream>>>(
      znb, W1T, b1, nullptr, h1, NTQ, 2 * DIMC, DIMC, 1);
  gemm_bf16_wmma_kernel<<<dim3(DIMC / 128, NTQ / 64), 256, 0, stream>>>(
      h1, W2T, b2, h2, nullptr, NTQ, DIMC, 2 * DIMC, 0);

  // 8) post-LN on (zn + h2), write transposed fp32 output
  final_ln_transpose_kernel<<<NTQ / 8, 256, 0, stream>>>(znf, h2, ln_post_g,
                                                         ln_post_b, out, NTQ);
}
